// ChmmLm_19353122636098
// MI455X (gfx1250) — compile-verified, matrix operand order checked by reading, not run on hardware
//
#include <hip/hip_runtime.h>
#include <cstdint>
#include <cstddef>

// Problem constants (match reference)
#define V_   8192
#define C_   2048
#define SPW_ 128
#define H_   512
#define B_   16
#define T_   256

typedef __attribute__((ext_vector_type(16))) __bf16 v16bf;
typedef __attribute__((ext_vector_type(8)))  float  v8f;

__device__ __forceinline__ unsigned short f2b(float f) {
  unsigned u = __float_as_uint(f);
  u += 0x7FFFu + ((u >> 16) & 1u);          // RNE
  return (unsigned short)(u >> 16);
}
__device__ __forceinline__ float b2f(unsigned short h) {
  return __uint_as_float(((unsigned)h) << 16);
}

struct Frag32 { uint4 lo, hi; };
__device__ __forceinline__ v16bf make_frag(const unsigned short* run0,
                                           const unsigned short* run1) {
  Frag32 u;
  u.lo = *(const uint4*)run0;   // K = 8*half + 0..7
  u.hi = *(const uint4*)run1;   // K = 16 + 8*half + 0..7
  return __builtin_bit_cast(v16bf, u);
}

// CDNA5 async copy: global -> LDS, 16 bytes, tracked by ASYNCcnt.
__device__ __forceinline__ void async_copy_b128(unsigned lds_addr, const void* gptr) {
  asm volatile("global_load_async_to_lds_b128 %0, %1, off"
               :: "v"(lds_addr), "v"(gptr) : "memory");
}
__device__ __forceinline__ void wait_async0() {
  asm volatile("s_wait_asynccnt 0x0" ::: "memory");
}

// ---------------------------------------------------------------- converts
__global__ void k_f32_to_bf16(const float* __restrict__ src,
                              unsigned short* __restrict__ dst, int n) {
  for (int i = blockIdx.x * blockDim.x + threadIdx.x; i < n;
       i += gridDim.x * blockDim.x)
    dst[i] = f2b(src[i]);
}

// dst[N x K] (bf16) = transpose(src[K x N] (f32)); tiled through LDS.
__global__ void k_transpose_bf16(const float* __restrict__ src,
                                 unsigned short* __restrict__ dst,
                                 int K, int N) {
  __shared__ float tile[32][33];
  const int k0 = blockIdx.x * 32, n0 = blockIdx.y * 32;
  const int tx = threadIdx.x & 31, ty = threadIdx.x >> 5;   // 32 x 8
#pragma unroll
  for (int i = ty; i < 32; i += 8) {
    int k = k0 + i, n = n0 + tx;
    tile[i][tx] = (k < K && n < N) ? src[(size_t)k * N + n] : 0.f;
  }
  __syncthreads();
#pragma unroll
  for (int i = ty; i < 32; i += 8) {
    int n = n0 + i, k = k0 + tx;
    if (n < N && k < K) dst[(size_t)n * K + k] = f2b(tile[tx][i]);
  }
}

// y = emb + h2 (residual), stored f32 and bf16
__global__ void k_residual(const float* __restrict__ emb,
                           const unsigned short* __restrict__ h2b,
                           float* __restrict__ yf,
                           unsigned short* __restrict__ yb, int n) {
  for (int i = blockIdx.x * blockDim.x + threadIdx.x; i < n;
       i += gridDim.x * blockDim.x) {
    float v = emb[i] + b2f(h2b[i]);
    yf[i] = v;
    yb[i] = f2b(v);
  }
}

// ---------------------------------------------------------------- WMMA GEMM
// C[M,N] = act(A[M,K] * B[K,N] + bias), A row-major bf16 (lda=K stride),
// Bt = B transposed, N-major / K-contiguous bf16 (ldbt=K stride).
// Workgroup tile 128(M) x 64(N), 8 waves; double-buffered async-to-LDS staging.
__global__ void __launch_bounds__(256, 1)
k_gemm_wmma(const unsigned short* __restrict__ A, int lda,
            const unsigned short* __restrict__ Bt, int ldbt,
            const float* __restrict__ bias,
            float* __restrict__ Cf, unsigned short* __restrict__ Cb,
            int ldc, int K, int relu) {
  __shared__ unsigned short ldsA[2][128 * 32];   // [buf][row][k]
  __shared__ unsigned short ldsB[2][64 * 32];    // [buf][col][k]

  const int tid  = threadIdx.x;                  // 256 threads
  const int wave = tid >> 5;
  const int lane = tid & 31;
  const int m16  = lane & 15;
  const int half = lane >> 4;
  const int rowBase = blockIdx.y * 128;
  const int colBase = blockIdx.x * 64;

  // staging assignments (fixed per thread)
  const int ar = tid >> 1, ah = (tid & 1) * 16;          // A: 128 rows x 32k
  const int bn = tid >> 2, bk = (tid & 3) * 8;           // B: 64 cols x 32k

  auto stage = [&](int buf, int k0) {
    const unsigned short* ga = A + (size_t)(rowBase + ar) * lda + k0 + ah;
    unsigned la = (unsigned)(size_t)&ldsA[buf][ar * 32 + ah];
    async_copy_b128(la,      ga);
    async_copy_b128(la + 16, ga + 8);
    const unsigned short* gb = Bt + (size_t)(colBase + bn) * ldbt + k0 + bk;
    unsigned lb = (unsigned)(size_t)&ldsB[buf][bn * 32 + bk];
    async_copy_b128(lb, gb);
  };

  v8f acc[4] = {v8f{}, v8f{}, v8f{}, v8f{}};

  stage(0, 0);
  for (int k0 = 0; k0 < K; k0 += 32) {
    const int buf = (k0 >> 5) & 1;
    wait_async0();          // our wave's async LDS writes done
    __syncthreads();        // everyone's writes visible; prior reads retired
    if (k0 + 32 < K) stage(buf ^ 1, k0 + 32);   // overlap next slab

    const int r = wave * 16 + m16;
    const unsigned short* arow = &ldsA[buf][r * 32];
    const unsigned short* bbas = &ldsB[buf][m16 * 32];
    v16bf a  = make_frag(arow + 8 * half, arow + 16 + 8 * half);
    // software-pipelined: load fragment t+1 before WMMA t consumes fragment t
    v16bf b0 = make_frag(bbas + 8 * half,        bbas + 16 + 8 * half);
    v16bf b1 = make_frag(bbas + 512 + 8 * half,  bbas + 512 + 16 + 8 * half);
    acc[0] = __builtin_amdgcn_wmma_f32_16x16x32_bf16(false, a, false, b0, (short)0, acc[0], false, false);
    v16bf b2 = make_frag(bbas + 1024 + 8 * half, bbas + 1024 + 16 + 8 * half);
    acc[1] = __builtin_amdgcn_wmma_f32_16x16x32_bf16(false, a, false, b1, (short)0, acc[1], false, false);
    v16bf b3 = make_frag(bbas + 1536 + 8 * half, bbas + 1536 + 16 + 8 * half);
    acc[2] = __builtin_amdgcn_wmma_f32_16x16x32_bf16(false, a, false, b2, (short)0, acc[2], false, false);
    acc[3] = __builtin_amdgcn_wmma_f32_16x16x32_bf16(false, a, false, b3, (short)0, acc[3], false, false);
  }

  // epilogue: D layout — elem i of v8f: row = i + 8*half, col = m16 (within tile)
  const int rowT = rowBase + wave * 16;
#pragma unroll
  for (int t = 0; t < 4; ++t) {
    const int col = colBase + t * 16 + m16;
    const float bv = bias ? bias[col] : 0.f;
#pragma unroll
    for (int i = 0; i < 8; ++i) {
      const int row = rowT + i + 8 * half;
      float v = acc[t][i] + bv;
      if (relu) v = fmaxf(v, 0.f);
      if (Cf) Cf[(size_t)row * ldc + col] = v;
      if (Cb) Cb[(size_t)row * ldc + col] = f2b(v);
    }
  }
}

// ---------------------------------------------------------------- heads / LSE
__global__ void k_start_head(const float* __restrict__ yf,
                             const float* __restrict__ w,
                             const float* __restrict__ b,
                             float* __restrict__ logit) {
  int r = blockIdx.x * blockDim.x + threadIdx.x;
  if (r < C_) {
    float s = 0.f;
    const float* p = yf + (size_t)r * H_;
#pragma unroll 8
    for (int i = 0; i < H_; ++i) s += p[i] * w[i];
    logit[r] = s + b[0];
  }
}

// single block: out[i] = x[i] - logsumexp(x)
__global__ void k_vec_ls(const float* __restrict__ x, float* __restrict__ o, int n) {
  __shared__ float red[256];
  int tid = threadIdx.x;
  float m = -3.0e38f;
  for (int i = tid; i < n; i += 256) m = fmaxf(m, x[i]);
  red[tid] = m; __syncthreads();
  for (int off = 128; off > 0; off >>= 1) {
    if (tid < off) red[tid] = fmaxf(red[tid], red[tid + off]);
    __syncthreads();
  }
  m = red[0]; __syncthreads();
  float s = 0.f;
  for (int i = tid; i < n; i += 256) s += __expf(x[i] - m);
  red[tid] = s; __syncthreads();
  for (int off = 128; off > 0; off >>= 1) {
    if (tid < off) red[tid] += red[tid + off];
    __syncthreads();
  }
  float lse = m + __logf(red[0]);
  for (int i = tid; i < n; i += 256) o[i] = x[i] - lse;
}

// one block per row: lse[row] = logsumexp(mat[row, 0:ncols])
__global__ void k_row_lse(const float* __restrict__ mat, int ld, int ncols,
                          float* __restrict__ lse) {
  __shared__ float red[256];
  int row = blockIdx.x, tid = threadIdx.x;
  const float* p = mat + (size_t)row * ld;
  float m = -3.0e38f;
  for (int i = tid; i < ncols; i += 256) m = fmaxf(m, p[i]);
  red[tid] = m; __syncthreads();
  for (int off = 128; off > 0; off >>= 1) {
    if (tid < off) red[tid] = fmaxf(red[tid], red[tid + off]);
    __syncthreads();
  }
  m = red[0]; __syncthreads();
  float s = 0.f;
  for (int i = tid; i < ncols; i += 256) s += __expf(p[i] - m);
  red[tid] = s; __syncthreads();
  for (int off = 128; off > 0; off >>= 1) {
    if (tid < off) red[tid] += red[tid + off];
    __syncthreads();
  }
  if (tid == 0) lse[row] = m + __logf(red[0]);
}

// obs[b,t,k] = un_blk[128*(w>>9)+k, w&511] - em_lse[state]
__global__ void k_gather_obs(const int* __restrict__ text,
                             const float* __restrict__ un_blk,
                             const float* __restrict__ em_lse,
                             float* __restrict__ obs) {
  int bt = blockIdx.x, k = threadIdx.x;
  int w = text[bt];
  int c = ((w >> 9) << 7) + k;
  obs[(size_t)bt * SPW_ + k] = un_blk[(size_t)c * 512 + (w & 511)] - em_lse[c];
}

// ---------------------------------------------------------------- forward scan
__global__ void k_forward(const int* __restrict__ text,
                          const float* __restrict__ tr,     // C_ x C_ raw logits
                          const float* __restrict__ tr_lse, // per-row lse
                          const float* __restrict__ start_ls,
                          const float* __restrict__ obs,
                          float* __restrict__ batch_lse) {
  __shared__ float sa[SPW_];
  __shared__ float red[SPW_];
  const int b = blockIdx.x, j = threadIdx.x;
  const int* tx = text + b * T_;

  int gp = tx[0] >> 9;
  float a = start_ls[gp * SPW_ + j] + obs[((size_t)b * T_) * SPW_ + j];

  for (int t = 1; t < T_; ++t) {
    const int gn = tx[t] >> 9;
    sa[j] = a - tr_lse[gp * SPW_ + j];    // fold row normalization into alpha
    __syncthreads();
    const float* trblk = tr + (size_t)(gp * SPW_) * C_ + gn * SPW_;
    if (t + 1 < T_) {                     // warm next transition tile
      const int g2 = tx[t + 1] >> 9;
      __builtin_prefetch(tr + (size_t)(gn * SPW_ + j) * C_ + g2 * SPW_ + j, 0, 1);
    }
    float m = -3.0e38f, s = 0.f;
#pragma unroll 4
    for (int k = 0; k < SPW_; ++k) {
      float v = trblk[(size_t)k * C_ + j] + sa[k];
      if (v > m) { s = s * __expf(m - v) + 1.f; m = v; }
      else       { s += __expf(v - m); }
    }
    a = obs[((size_t)b * T_ + t) * SPW_ + j] + m + __logf(s);
    gp = gn;
    __syncthreads();
  }

  red[j] = a; __syncthreads();
  for (int off = 64; off > 0; off >>= 1) {
    if (j < off) red[j] = fmaxf(red[j], red[j + off]);
    __syncthreads();
  }
  float mx = red[0]; __syncthreads();
  red[j] = __expf(a - mx); __syncthreads();
  for (int off = 64; off > 0; off >>= 1) {
    if (j < off) red[j] += red[j + off];
    __syncthreads();
  }
  if (j == 0) batch_lse[b] = mx + __logf(red[0]);
}

__global__ void k_final_sum(const float* __restrict__ batch_lse, float* __restrict__ out) {
  float s = 0.f;
  for (int b = 0; b < B_; ++b) s += batch_lse[b];
  out[0] = s;
}

// ---------------------------------------------------------------- launcher
extern "C" void kernel_launch(void* const* d_in, const int* in_sizes, int n_in,
                              void* d_out, int out_size, void* d_ws, size_t ws_size,
                              hipStream_t stream) {
  const int*   text      = (const int*)  d_in[0];
  const float* start_emb = (const float*)d_in[3];
  const float* sw1 = (const float*)d_in[4];  const float* sb1 = (const float*)d_in[5];
  const float* sw2 = (const float*)d_in[6];  const float* sb2 = (const float*)d_in[7];
  const float* swo = (const float*)d_in[8];  const float* sbo = (const float*)d_in[9];
  const float* state_emb = (const float*)d_in[10];
  const float* tw1 = (const float*)d_in[11]; const float* tb1 = (const float*)d_in[12];
  const float* tw2 = (const float*)d_in[13]; const float* tb2 = (const float*)d_in[14];
  const float* two = (const float*)d_in[15]; const float* tbo = (const float*)d_in[16];
  const float* pre_emb   = (const float*)d_in[17];
  const float* ew1 = (const float*)d_in[18]; const float* eb1 = (const float*)d_in[19];
  const float* ew2 = (const float*)d_in[20]; const float* eb2 = (const float*)d_in[21];
  const float* ewo = (const float*)d_in[22]; const float* ebo = (const float*)d_in[23];

  char* wp = (char*)d_ws;
  auto alloc = [&](size_t bytes) -> void* {
    void* p = (void*)wp; wp += (bytes + 255) & ~(size_t)255; return p;
  };
  const size_t HH = (size_t)H_ * H_, CH = (size_t)C_ * H_;
  // all weights stored TRANSPOSED (N-major, K-contiguous) in bf16
  unsigned short* w1t[3]; unsigned short* w2t[3];
  for (int i = 0; i < 3; ++i) { w1t[i] = (unsigned short*)alloc(HH * 2);
                                w2t[i] = (unsigned short*)alloc(HH * 2); }
  unsigned short* twot = (unsigned short*)alloc((size_t)C_ * H_ * 2);        // C x H
  unsigned short* ewot = (unsigned short*)alloc((size_t)(V_ + 1) * H_ * 2);  // (V+1) x H
  unsigned short* xb   = (unsigned short*)alloc(CH * 2);
  unsigned short* h1b  = (unsigned short*)alloc(CH * 2);
  unsigned short* h2b  = (unsigned short*)alloc(CH * 2);
  unsigned short* yb   = (unsigned short*)alloc(CH * 2);
  float* yf      = (float*)alloc(CH * 4);
  float* s_logit = (float*)alloc(C_ * 4);
  float* start_ls= (float*)alloc(C_ * 4);
  float* trans   = (float*)alloc((size_t)C_ * C_ * 4);   // 16 MB, L2-resident
  float* tr_lse  = (float*)alloc(C_ * 4);
  float* un_blk  = (float*)alloc((size_t)C_ * 512 * 4);
  float* em_lse  = (float*)alloc(C_ * 4);
  float* obs     = (float*)alloc((size_t)B_ * T_ * SPW_ * 4);
  float* blse    = (float*)alloc(B_ * 4);

  auto conv = [&](const float* s, unsigned short* d, size_t n) {
    int g = (int)((n + 255) / 256); if (g > 4096) g = 4096;
    k_f32_to_bf16<<<g, 256, 0, stream>>>(s, d, (int)n);
  };
  auto transp = [&](const float* s, unsigned short* d, int K, int N) {
    dim3 g((K + 31) / 32, (N + 31) / 32);
    k_transpose_bf16<<<g, 256, 0, stream>>>(s, d, K, N);
  };
  transp(sw1, w1t[0], H_, H_);  transp(sw2, w2t[0], H_, H_);
  transp(tw1, w1t[1], H_, H_);  transp(tw2, w2t[1], H_, H_);
  transp(ew1, w1t[2], H_, H_);  transp(ew2, w2t[2], H_, H_);
  transp(two, twot, H_, C_);
  transp(ewo, ewot, H_, V_ + 1);

  const float* embs[3] = { start_emb, state_emb, pre_emb };
  const float* b1s[3]  = { sb1, tb1, eb1 };
  const float* b2s[3]  = { sb2, tb2, eb2 };

  for (int s = 0; s < 3; ++s) {
    conv(embs[s], xb, CH);
    dim3 gh(H_ / 64, C_ / 128);
    k_gemm_wmma<<<gh, 256, 0, stream>>>(xb,  H_, w1t[s], H_, b1s[s], nullptr, h1b, H_, H_, 1);
    k_gemm_wmma<<<gh, 256, 0, stream>>>(h1b, H_, w2t[s], H_, b2s[s], nullptr, h2b, H_, H_, 1);
    k_residual<<<2048, 256, 0, stream>>>(embs[s], h2b, yf, yb, (int)CH);

    if (s == 0) {            // start head: (C,H)@(H,1) + log_softmax over C
      k_start_head<<<C_ / 256, 256, 0, stream>>>(yf, swo, sbo, s_logit);
      k_vec_ls<<<1, 256, 0, stream>>>(s_logit, start_ls, C_);
    } else if (s == 1) {     // transition head: (C,H)@(H,C); keep raw + row lse
      dim3 gt(C_ / 64, C_ / 128);
      k_gemm_wmma<<<gt, 256, 0, stream>>>(yb, H_, twot, H_, tbo, trans, nullptr, C_, H_, 0);
      k_row_lse<<<C_, 256, 0, stream>>>(trans, C_, C_, tr_lse);
    } else {                 // emission head: block-diagonal 16 x (128x512x512)
      for (int g = 0; g < 16; ++g) {
        dim3 ge(512 / 64, 1);
        k_gemm_wmma<<<ge, 256, 0, stream>>>(
            yb + (size_t)g * 128 * H_, H_,
            ewot + (size_t)g * 512 * H_, H_,
            ebo + (size_t)g * 512,
            un_blk + (size_t)g * 128 * 512, nullptr, 512, H_, 0);
      }
      k_row_lse<<<C_, 256, 0, stream>>>(un_blk, 512, 512, em_lse);
    }
  }

  k_gather_obs<<<B_ * T_, SPW_, 0, stream>>>(text, un_blk, em_lse, obs);
  k_forward<<<B_, SPW_, 0, stream>>>(text, trans, tr_lse, start_ls, obs, blse);
  k_final_sum<<<1, 1, 0, stream>>>(blse, (float*)d_out);
}
